// RSSM_84018150244965
// MI455X (gfx1250) — matrix operand order repaired and checked by reading, not compiled, not run on hardware
//
#include <hip/hip_runtime.h>
#include <hip/hip_bf16.h>
#include <math.h>

// ---------------- problem dims ----------------
#define Bb 32
#define Tt 64
#define Aa 32
#define Ee 1536

#define K_II   1056
#define K_IIP  1088   // zero-padded to a multiple of 64
#define K_GRU  2048
#define K_IO   1024
#define K_ST   1024
#define K_OO   2560

#define N_II 1024
#define N_GRU 3072
#define N_IO 1024
#define N_ST 1024
#define N_OO 1024

// ---------------- launch shape ----------------
#define G_WG 16       // cooperative workgroups (all resident on MI455X)
#define BLK 256       // 8 wave32 per WG
#define WPW 8
#define AS_STRIDE 72  // LDS A row stride in bf16 (64 + 8 pad; 144B keeps 16B alignment)
#define ASBUF (32 * AS_STRIDE)  // one A panel buffer (32 rows x 64 k)

typedef __attribute__((ext_vector_type(16))) __bf16 bf16x16;
typedef __attribute__((ext_vector_type(8)))  float  f32x8;

// ---------------- workspace layout (bytes) ----------------
constexpr size_t OFF_SYNC   = 0;                              // 2 u32, padded
constexpr size_t OFF_WT_II  = 256;
constexpr size_t SZ_WT_II   = (size_t)N_II  * K_IIP * 2;
constexpr size_t OFF_WT_GRU = OFF_WT_II  + SZ_WT_II;
constexpr size_t SZ_WT_GRU  = (size_t)N_GRU * K_GRU * 2;
constexpr size_t OFF_WT_IO  = OFF_WT_GRU + SZ_WT_GRU;
constexpr size_t SZ_WT_IO   = (size_t)N_IO  * K_IO  * 2;
constexpr size_t OFF_WT_ST  = OFF_WT_IO  + SZ_WT_IO;
constexpr size_t SZ_WT_ST   = (size_t)N_ST  * K_ST  * 2;
constexpr size_t OFF_WT_OO  = OFF_WT_ST  + SZ_WT_ST;
constexpr size_t SZ_WT_OO   = (size_t)N_OO  * K_OO  * 2;
constexpr size_t OFF_WT_OS  = OFF_WT_OO  + SZ_WT_OO;
constexpr size_t SZ_WT_OS   = (size_t)N_ST  * K_ST  * 2;
constexpr size_t OFF_DETER  = OFF_WT_OS  + SZ_WT_OS;          // f32 [32][1024]
constexpr size_t OFF_STOCH  = OFF_DETER  + 32 * 1024 * 4;     // f32 [32][1024]
constexpr size_t OFF_ZI     = OFF_STOCH  + 32 * 1024 * 4;     // f32 [32][1024]
constexpr size_t OFF_ZG     = OFF_ZI     + 32 * 1024 * 4;     // f32 [32][3072]
constexpr size_t OFF_ZO     = OFF_ZG     + 32 * 3072 * 4;     // f32 [32][1024]
constexpr size_t OFF_ZY     = OFF_ZO     + 32 * 1024 * 4;     // f32 [32][1024]
constexpr size_t OFF_PART_I = OFF_ZY     + 32 * 1024 * 4;     // float2 [16][32]
constexpr size_t OFF_PART_G = OFF_PART_I + 16 * 32 * 8;
constexpr size_t OFF_PART_O = OFF_PART_G + 16 * 32 * 8;       // float2 [8][32]
constexpr size_t OFF_PART_Y = OFF_PART_O + 8 * 32 * 8;
constexpr size_t WS_BYTES   = OFF_PART_Y + 8 * 32 * 8;        // ~27.4 MB total

struct Args {
  const float *action, *embed, *is_first, *gum_prior, *gum_post;
  const float *W_img_in, *b_img_in, *s_img_in, *o_img_in;
  const float *W_gru, *b_gru, *s_gru, *o_gru;
  const float *W_img_out, *b_img_out, *s_img_out, *o_img_out;
  const float *W_img_stats, *b_img_stats;
  const float *W_obs_out, *b_obs_out, *s_obs_out, *o_obs_out;
  const float *W_obs_stats, *b_obs_stats;
  float* out;
  char*  ws;
};

// ---------------- grid-wide barrier (generation counter in L2) ----------------
__device__ __forceinline__ void grid_sync(unsigned* cnt, unsigned* gen) {
  __threadfence();            // make this WG's global writes visible device-wide
  __syncthreads();
  if (threadIdx.x == 0) {
    unsigned g = __hip_atomic_load(gen, __ATOMIC_ACQUIRE, __HIP_MEMORY_SCOPE_AGENT);
    unsigned arrived = __hip_atomic_fetch_add(cnt, 1u, __ATOMIC_ACQ_REL,
                                              __HIP_MEMORY_SCOPE_AGENT) + 1u;
    if (arrived == (unsigned)G_WG) {
      __hip_atomic_store(cnt, 0u, __ATOMIC_RELAXED, __HIP_MEMORY_SCOPE_AGENT);
      __hip_atomic_fetch_add(gen, 1u, __ATOMIC_ACQ_REL, __HIP_MEMORY_SCOPE_AGENT);
    } else {
      while (__hip_atomic_load(gen, __ATOMIC_ACQUIRE, __HIP_MEMORY_SCOPE_AGENT) == g) {
        __builtin_amdgcn_s_sleep(1);
      }
    }
  }
  __syncthreads();
}

__device__ __forceinline__ float sigmoidf_(float x) { return 1.f / (1.f + expf(-x)); }
__device__ __forceinline__ float siluf_(float x)    { return x / (1.f + expf(-x)); }

// LN mean / rstd for 32 rows from per-WG partial (sum, sumsq)
__device__ __forceinline__ void compute_meanvar(const float2* part, int npart, float Nf,
                                                float* ldsMean, float* ldsRstd) {
  if (threadIdx.x < 32) {
    float S = 0.f, Q = 0.f;
    for (int g = 0; g < npart; ++g) {
      float2 p = part[g * 32 + threadIdx.x];
      S += p.x; Q += p.y;
    }
    float mean = S / Nf;
    float var  = Q / Nf - mean * mean;
    ldsMean[threadIdx.x] = mean;
    ldsRstd[threadIdx.x] = rsqrtf(var + 1e-3f);
  }
  __syncthreads();
}

// ---------------- generic column-partitioned GEMM phase ----------------
// D = A(32 x K, bf16 staged via loadA) * Wt^T + bias ; Wt is bf16 [N][K], K % 64 == 0.
// NT is the exact per-wave tile count so accumulators are compile-time indexed
// and stay in VGPRs. LDS A panel is double-buffered: chunk c+1 is staged while
// WMMAs consume chunk c -> one workgroup barrier per 64-K chunk.
template <int NT, class LoadAFn>
__device__ __forceinline__ void run_matmul(
    const __bf16* __restrict__ Wt, const int KDIM, const int NDIM,
    const float* __restrict__ bias,
    float* __restrict__ zbuf, float* __restrict__ statsout, const int t,
    float2* __restrict__ my_part,
    const int gwave, const int gnw,
    __bf16* Asub0, float* redS, float* redQ, LoadAFn loadA) {
  const int tid  = threadIdx.x;
  const int lane = tid & 31;
  const int half = lane >> 4;   // K-half selector (wave32 layout)
  const int nl   = lane & 15;

  f32x8 acc[NT];
  int mtv[NT];
  const __bf16* brow_base[NT];
#pragma unroll
  for (int ti = 0; ti < NT; ++ti) {
    const int tx = gwave + ti * gnw;       // exact partition: always < ntiles
    mtv[ti] = tx & 1;
    brow_base[ti] = Wt + (size_t)((tx >> 1) * 16 + nl) * KDIM + half * 16;
#pragma unroll
    for (int i = 0; i < 8; ++i) acc[ti][i] = 0.f;
  }
  if (zbuf && tid < 32) { redS[tid] = 0.f; redQ[tid] = 0.f; }

  const int nchunk = KDIM >> 6;     // 64-K chunks (two WMMA k-steps each)
  const int sm = tid >> 3;          // staging row 0..31
  const int sk = (tid & 7) << 3;    // staging col 0..56 step 8

  auto stage = [&](int c, __bf16* dstbuf) {
    float v[8];
    loadA(sm, (c << 6) + sk, v);
    union { uint4 q; __bf16 h[8]; } u;
#pragma unroll
    for (int i = 0; i < 8; ++i) u.h[i] = (__bf16)v[i];
    *(uint4*)(dstbuf + sm * AS_STRIDE + sk) = u.q;   // one 16B LDS store
  };

  stage(0, Asub0);
  __syncthreads();
  for (int c = 0; c < nchunk; ++c) {
    __bf16* cur = Asub0 + ((c & 1) ? ASBUF : 0);
    __bf16* nxt = Asub0 + ((c & 1) ? 0 : ASBUF);
    if (c + 1 < nchunk) stage(c + 1, nxt);   // overlap staging with WMMA below
#pragma unroll
    for (int ti = 0; ti < NT; ++ti) {
      const __bf16* brow = brow_base[ti] + (c << 6);
      union { bf16x16 v; uint4 q[2]; } ub0, ub1, ua0, ua1;
      // B frags for both k-steps first (contiguous 32B each, long load clause)
      ub0.q[0] = ((const uint4*)brow)[0];
      ub0.q[1] = ((const uint4*)brow)[1];
      ub1.q[0] = ((const uint4*)(brow + 32))[0];
      ub1.q[1] = ((const uint4*)(brow + 32))[1];
      __builtin_prefetch(brow + 64, 0, 1);  // global_prefetch next chunk of this row
      // A frags: lane nl = row M; e=0..15 -> k=half*16+e
      const __bf16* arow = cur + (mtv[ti] * 16 + nl) * AS_STRIDE;
      ua0.q[0] = *(const uint4*)(arow + half * 16);
      ua0.q[1] = *(const uint4*)(arow + half * 16 + 8);
      ua1.q[0] = *(const uint4*)(arow + 32 + half * 16);
      ua1.q[1] = *(const uint4*)(arow + 40 + half * 16);
      acc[ti] = __builtin_amdgcn_wmma_f32_16x16x32_bf16(
          false, ua0.v, false, ub0.v, (short)0, acc[ti], false, false);
      acc[ti] = __builtin_amdgcn_wmma_f32_16x16x32_bf16(
          false, ua1.v, false, ub1.v, (short)0, acc[ti], false, false);
    }
    __syncthreads();   // everyone done reading cur; nxt fully staged
  }

  // epilogue: bias add, store, LN partial row sums via wave32 shfl butterflies
#pragma unroll
  for (int ti = 0; ti < NT; ++ti) {
    const int tx = gwave + ti * gnw;
    const int n  = (tx >> 1) * 16 + nl;
    const float bn = bias[n];
#pragma unroll
    for (int i = 0; i < 8; ++i) {
      const int m = mtv[ti] * 16 + half * 8 + i;   // C/D layout: M = i + 8*half
      float v = acc[ti][i] + bn;
      if (zbuf) {
        zbuf[(size_t)m * NDIM + n] = v;
        float s = v, q = v * v;
#pragma unroll
        for (int mk = 1; mk < 16; mk <<= 1) {
          s += __shfl_xor(s, mk, 32);
          q += __shfl_xor(q, mk, 32);
        }
        if (nl == 0) { atomicAdd(&redS[m], s); atomicAdd(&redQ[m], q); }
      } else {
        statsout[((size_t)m * Tt + t) * NDIM + n] = v;
      }
    }
  }
  __syncthreads();
  if (zbuf && tid < 32) my_part[tid] = make_float2(redS[tid], redQ[tid]);
}

// ---------------- prep: bf16-transpose weights once per call + reset barrier ----------
__global__ void rssm_prep_kernel(Args a) {
  const size_t idx0   = (size_t)blockIdx.x * blockDim.x + threadIdx.x;
  const size_t stride = (size_t)gridDim.x * blockDim.x;
  if (idx0 == 0) {
    unsigned* s = (unsigned*)(a.ws + OFF_SYNC);
    s[0] = 0u; s[1] = 0u;
  }
  const float* src[6] = {a.W_img_in, a.W_gru, a.W_img_out, a.W_img_stats, a.W_obs_out, a.W_obs_stats};
  __bf16* dst[6] = {(__bf16*)(a.ws + OFF_WT_II), (__bf16*)(a.ws + OFF_WT_GRU),
                    (__bf16*)(a.ws + OFF_WT_IO), (__bf16*)(a.ws + OFF_WT_ST),
                    (__bf16*)(a.ws + OFF_WT_OO), (__bf16*)(a.ws + OFF_WT_OS)};
  const int Kpad[6] = {K_IIP, K_GRU, K_IO, K_ST, K_OO, K_ST};  // padded (stored) K
  const int Ksrc[6] = {K_II,  K_GRU, K_IO, K_ST, K_OO, K_ST};  // true K
  const int Ns[6]   = {N_II,  N_GRU, N_IO, N_ST, N_OO, N_ST};
  for (int mtx = 0; mtx < 6; ++mtx) {
    const size_t tot = (size_t)Kpad[mtx] * Ns[mtx];
    for (size_t i = idx0; i < tot; i += stride) {
      const size_t n = i / Kpad[mtx], k = i % Kpad[mtx];
      dst[mtx][i] = (k < (size_t)Ksrc[mtx]) ? (__bf16)src[mtx][k * Ns[mtx] + n] : (__bf16)0.f;
    }
  }
}

// ---------------- persistent cooperative RSSM scan ----------------
__global__ void __launch_bounds__(BLK, 1) rssm_scan_kernel(Args a) {
  __shared__ __attribute__((aligned(16))) __bf16 Asub[2 * ASBUF];  // double-buffered A panel
  __shared__ float ldsMean[32], ldsRstd[32], redS[32], redQ[32];

  unsigned* syncw = (unsigned*)(a.ws + OFF_SYNC);
  unsigned* cnt = syncw, * gen = syncw + 1;
  const __bf16* Wt_ii = (const __bf16*)(a.ws + OFF_WT_II);
  const __bf16* Wt_gr = (const __bf16*)(a.ws + OFF_WT_GRU);
  const __bf16* Wt_io = (const __bf16*)(a.ws + OFF_WT_IO);
  const __bf16* Wt_st = (const __bf16*)(a.ws + OFF_WT_ST);
  const __bf16* Wt_oo = (const __bf16*)(a.ws + OFF_WT_OO);
  const __bf16* Wt_os = (const __bf16*)(a.ws + OFF_WT_OS);
  float* deter = (float*)(a.ws + OFF_DETER);
  float* stoch = (float*)(a.ws + OFF_STOCH);
  float* zi = (float*)(a.ws + OFF_ZI);
  float* zg = (float*)(a.ws + OFF_ZG);
  float* zo = (float*)(a.ws + OFF_ZO);
  float* zy = (float*)(a.ws + OFF_ZY);
  float2* part_i = (float2*)(a.ws + OFF_PART_I);
  float2* part_g = (float2*)(a.ws + OFF_PART_G);
  float2* part_o = (float2*)(a.ws + OFF_PART_O);
  float2* part_y = (float2*)(a.ws + OFF_PART_Y);

  float* out_det  = a.out;                                   // [B,T,1024]
  float* out_qlog = a.out + (size_t)Bb * Tt * 1024;          // post_logit
  float* out_qst  = a.out + (size_t)2 * Bb * Tt * 1024;      // post_stoch
  float* out_plog = a.out + (size_t)3 * Bb * Tt * 1024;      // prior_logit
  float* out_pst  = a.out + (size_t)4 * Bb * Tt * 1024;      // prior_stoch

  const int wg  = blockIdx.x;
  const int tid = threadIdx.x;
  const int wv  = tid >> 5;
  const int gt  = wg * BLK + tid;
  const int gwave_all = wg * WPW + wv;          // wave id across all 16 WGs (128 waves)

  // P0: zero carried state (harness does not re-init workspace between replays)
  for (int i = gt; i < 32 * 1024; i += G_WG * BLK) { deter[i] = 0.f; stoch[i] = 0.f; }
  grid_sync(cnt, gen);

  for (int t = 0; t < Tt; ++t) {
    // ---- P1: img_in  z = [stoch*m, action, 0pad] @ W + b (128 tiles / 128 waves, NT=1)
    {
      auto loadA = [&](int m, int k0, float* v) {
        if (k0 < 1024) {
          const float msk = 1.f - a.is_first[m * Tt + t];
          const float* p = stoch + m * 1024 + k0;
#pragma unroll
          for (int i = 0; i < 8; ++i) v[i] = p[i] * msk;
        } else if (k0 < 1056) {
          const float* p = a.action + ((size_t)m * Tt + t) * Aa + (k0 - 1024);
#pragma unroll
          for (int i = 0; i < 8; ++i) v[i] = p[i];
        } else {
#pragma unroll
          for (int i = 0; i < 8; ++i) v[i] = 0.f;   // K padding
        }
      };
      run_matmul<1>(Wt_ii, K_IIP, N_II, a.b_img_in, zi, nullptr, t,
                    part_i + wg * 32, gwave_all, G_WG * WPW,
                    Asub, redS, redQ, loadA);
    }
    grid_sync(cnt, gen);

    // ---- P2: GRU  z = [deter*m, silu(ln(zi))] @ W + b (384 tiles / 128 waves, NT=3)
    compute_meanvar(part_i, G_WG, (float)N_II, ldsMean, ldsRstd);
    {
      auto loadA = [&](int m, int k0, float* v) {
        if (k0 < 1024) {
          const float msk = 1.f - a.is_first[m * Tt + t];
          const float* p = deter + m * 1024 + k0;
#pragma unroll
          for (int i = 0; i < 8; ++i) v[i] = p[i] * msk;
        } else {
          const int kk = k0 - 1024;
          const float mean = ldsMean[m], rs = ldsRstd[m];
#pragma unroll
          for (int i = 0; i < 8; ++i) {
            float u = (zi[m * 1024 + kk + i] - mean) * rs * a.s_img_in[kk + i] + a.o_img_in[kk + i];
            v[i] = siluf_(u);
          }
        }
      };
      run_matmul<3>(Wt_gr, K_GRU, N_GRU, a.b_gru, zg, nullptr, t,
                    part_g + wg * 32, gwave_all, G_WG * WPW,
                    Asub, redS, redQ, loadA);
    }
    grid_sync(cnt, gen);

    // ---- P2.5: GRU elementwise deter update (column-partitioned, in-place)
    compute_meanvar(part_g, G_WG, (float)N_GRU, ldsMean, ldsRstd);
    {
      const int e0 = gt * 8;   // 4096 threads x 8 = 32768 = 32*1024
      const int b = e0 >> 10, d = e0 & 1023;
      const float mean = ldsMean[b], rs = ldsRstd[b];
      const float msk = 1.f - a.is_first[b * Tt + t];
#pragma unroll
      for (int i = 0; i < 8; ++i) {
        const int dd = d + i;
        float r  = (zg[b * 3072 + dd]        - mean) * rs * a.s_gru[dd]        + a.o_gru[dd];
        float cn = (zg[b * 3072 + 1024 + dd] - mean) * rs * a.s_gru[1024 + dd] + a.o_gru[1024 + dd];
        float up = (zg[b * 3072 + 2048 + dd] - mean) * rs * a.s_gru[2048 + dd] + a.o_gru[2048 + dd];
        r  = sigmoidf_(r);
        cn = tanhf(r * cn);
        up = sigmoidf_(up - 1.f);
        const float dn = up * cn + (1.f - up) * deter[b * 1024 + dd] * msk;
        deter[b * 1024 + dd] = dn;
        out_det[((size_t)b * Tt + t) * 1024 + dd] = dn;
      }
    }
    grid_sync(cnt, gen);

    // ---- P3: img_out (WGs 0-7) || obs_out (WGs 8-15), 128 tiles / 64 waves, NT=2
    if (wg < 8) {
      auto loadA = [&](int m, int k0, float* v) {
        const float* p = deter + m * 1024 + k0;
#pragma unroll
        for (int i = 0; i < 8; ++i) v[i] = p[i];
      };
      run_matmul<2>(Wt_io, K_IO, N_IO, a.b_img_out, zo, nullptr, t,
                    part_o + wg * 32, wg * WPW + wv, 8 * WPW,
                    Asub, redS, redQ, loadA);
    } else {
      auto loadA = [&](int m, int k0, float* v) {
        if (k0 < 1024) {
          const float* p = deter + m * 1024 + k0;
#pragma unroll
          for (int i = 0; i < 8; ++i) v[i] = p[i];
        } else {
          const float* p = a.embed + ((size_t)m * Tt + t) * Ee + (k0 - 1024);
#pragma unroll
          for (int i = 0; i < 8; ++i) v[i] = p[i];
        }
      };
      run_matmul<2>(Wt_oo, K_OO, N_OO, a.b_obs_out, zy, nullptr, t,
                    part_y + (wg - 8) * 32, (wg - 8) * WPW + wv, 8 * WPW,
                    Asub, redS, redQ, loadA);
    }
    grid_sync(cnt, gen);

    // ---- P4: stats heads, logits straight into d_out (NT=2)
    if (wg < 8) {
      compute_meanvar(part_o, 8, (float)N_IO, ldsMean, ldsRstd);
      auto loadA = [&](int m, int k0, float* v) {
        const float mean = ldsMean[m], rs = ldsRstd[m];
#pragma unroll
        for (int i = 0; i < 8; ++i) {
          float u = (zo[m * 1024 + k0 + i] - mean) * rs * a.s_img_out[k0 + i] + a.o_img_out[k0 + i];
          v[i] = siluf_(u);
        }
      };
      run_matmul<2>(Wt_st, K_ST, N_ST, a.b_img_stats, nullptr, out_plog, t,
                    nullptr, wg * WPW + wv, 8 * WPW,
                    Asub, redS, redQ, loadA);
    } else {
      compute_meanvar(part_y, 8, (float)N_OO, ldsMean, ldsRstd);
      auto loadA = [&](int m, int k0, float* v) {
        const float mean = ldsMean[m], rs = ldsRstd[m];
#pragma unroll
        for (int i = 0; i < 8; ++i) {
          float u = (zy[m * 1024 + k0 + i] - mean) * rs * a.s_obs_out[k0 + i] + a.o_obs_out[k0 + i];
          v[i] = siluf_(u);
        }
      };
      run_matmul<2>(Wt_os, K_ST, N_ST, a.b_obs_stats, nullptr, out_qlog, t,
                    nullptr, (wg - 8) * WPW + wv, 8 * WPW,
                    Asub, redS, redQ, loadA);
    }
    grid_sync(cnt, gen);

    // ---- P5: Gumbel-max one-hots (2048 (which,b,s) tasks over 4096 threads)
    if (gt < 2048) {
      const int which = gt >> 10;          // 0 = post, 1 = prior
      const int bs = gt & 1023;
      const int b = bs >> 5, s = bs & 31;
      const size_t lbase = ((size_t)b * Tt + t) * 1024 + s * 32;
      const size_t gbase = (((size_t)t * Bb + b) * 32 + s) * 32;
      const float* logit = (which == 0 ? out_qlog : out_plog) + lbase;
      const float* gum   = (which == 0 ? a.gum_post : a.gum_prior) + gbase;
      float* stout       = (which == 0 ? out_qst : out_pst) + lbase;
      int best = 0;
      float bv = -3.4e38f;
      for (int c = 0; c < 32; ++c) {
        float v = logit[c] + gum[c];
        if (v > bv) { bv = v; best = c; }   // strict '>' == jnp.argmax first-max
      }
      for (int c = 0; c < 32; ++c) {
        const float oh = (c == best) ? 1.f : 0.f;
        stout[c] = oh;
        if (which == 0) stoch[b * 1024 + s * 32 + c] = oh;  // carried posterior sample
      }
    }
    grid_sync(cnt, gen);
  }
}

extern "C" void kernel_launch(void* const* d_in, const int* in_sizes, int n_in,
                              void* d_out, int out_size, void* d_ws, size_t ws_size,
                              hipStream_t stream) {
  (void)in_sizes; (void)n_in; (void)out_size; (void)ws_size;  // needs ~28 MB of d_ws
  Args a;
  a.action      = (const float*)d_in[0];
  a.embed       = (const float*)d_in[1];
  a.is_first    = (const float*)d_in[2];
  a.gum_prior   = (const float*)d_in[3];
  a.gum_post    = (const float*)d_in[4];
  a.W_img_in    = (const float*)d_in[5];
  a.b_img_in    = (const float*)d_in[6];
  a.s_img_in    = (const float*)d_in[7];
  a.o_img_in    = (const float*)d_in[8];
  a.W_gru       = (const float*)d_in[9];
  a.b_gru       = (const float*)d_in[10];
  a.s_gru       = (const float*)d_in[11];
  a.o_gru       = (const float*)d_in[12];
  a.W_img_out   = (const float*)d_in[13];
  a.b_img_out   = (const float*)d_in[14];
  a.s_img_out   = (const float*)d_in[15];
  a.o_img_out   = (const float*)d_in[16];
  a.W_img_stats = (const float*)d_in[17];
  a.b_img_stats = (const float*)d_in[18];
  a.W_obs_out   = (const float*)d_in[19];
  a.b_obs_out   = (const float*)d_in[20];
  a.s_obs_out   = (const float*)d_in[21];
  a.o_obs_out   = (const float*)d_in[22];
  a.W_obs_stats = (const float*)d_in[23];
  a.b_obs_stats = (const float*)d_in[24];
  a.out = (float*)d_out;
  a.ws  = (char*)d_ws;

  rssm_prep_kernel<<<512, BLK, 0, stream>>>(a);     // bf16 weight transpose + barrier reset
  rssm_scan_kernel<<<G_WG, BLK, 0, stream>>>(a);    // persistent cooperative T-scan
}
